// AttentionBlock_36155034698411
// MI455X (gfx1250) — matrix operand order repaired and checked by reading, not compiled, not run on hardware
//
#include <hip/hip_runtime.h>
#include <hip/hip_bf16.h>
#include <stdint.h>

// Problem constants (match the JAX reference)
#define B_   4
#define C_   256
#define L_   4096      // H*W
#define NH_  4
#define HD_  64        // C/NH
#define G_   8

typedef __attribute__((ext_vector_type(16))) __bf16 bf16x16;
typedef __attribute__((ext_vector_type(8)))  float  f32x8;

union Frag16 { bf16x16 v; uint4 q[2]; };

__device__ __forceinline__ unsigned short f2bf(float f) {
  uint32_t u = __float_as_uint(f);
  u += 0x7FFFu + ((u >> 16) & 1u);   // round-to-nearest-even
  return (unsigned short)(u >> 16);
}

// gfx1250 async DMA: global -> LDS, 16 bytes per lane, tracked by ASYNCcnt.
__device__ __forceinline__ void async_b128(uint32_t lds_addr, uint64_t base, uint32_t off) {
  asm volatile("global_load_async_to_lds_b128 %0, %1, %2"
               :: "v"(lds_addr), "v"(off), "s"(base) : "memory");
}
template <int N>
__device__ __forceinline__ void wait_async() {
  asm volatile("s_wait_asynccnt %0" :: "n"(N) : "memory");
}
__device__ __forceinline__ uint32_t lds_off32(const void* p) {
  return (uint32_t)(uintptr_t)p;   // low 32 bits of generic ptr = LDS offset
}

// ---------------------------------------------------------------- weights -> bf16
__global__ void k_f32_to_bf16(const float* __restrict__ src,
                              unsigned short* __restrict__ dst, int n) {
  int i = blockIdx.x * blockDim.x + threadIdx.x;
  if (i < n) dst[i] = f2bf(src[i]);
}

// ---------------------------------------------------------------- GroupNorm -> bf16 h_t
// one block per (b, g); output TRANSPOSED: h_t[b][l][c]  ([N][K] layout for the GEMMs)
__global__ __launch_bounds__(256)
void k_groupnorm(const float* __restrict__ x, const float* __restrict__ gw,
                 const float* __restrict__ gb, unsigned short* __restrict__ ht) {
  const int bg = blockIdx.x;
  const int b = bg / G_, g = bg % G_;
  const int CG = C_ / G_;                     // 32
  const long base = ((long)b * C_ + (long)g * CG) * L_;
  const int n = CG * L_;                      // 131072
  float s = 0.f, s2 = 0.f;
  for (int i = threadIdx.x; i < n; i += 256) {
    float v = x[base + i];
    s += v; s2 += v * v;
  }
  __shared__ float r1[256], r2[256];
  r1[threadIdx.x] = s; r2[threadIdx.x] = s2;
  __syncthreads();
  for (int off = 128; off > 0; off >>= 1) {
    if (threadIdx.x < off) {
      r1[threadIdx.x] += r1[threadIdx.x + off];
      r2[threadIdx.x] += r2[threadIdx.x + off];
    }
    __syncthreads();
  }
  const float mean = r1[0] / (float)n;
  const float var  = r2[0] / (float)n - mean * mean;
  const float rstd = rsqrtf(var + 1e-5f);
  for (int i = threadIdx.x; i < n; i += 256) {
    const int c = g * CG + (i >> 12);         // i / L_
    const int l = i & (L_ - 1);
    float v = (x[base + i] - mean) * rstd * gw[c] + gb[c];
    ht[((long)b * L_ + l) * C_ + c] = f2bf(v);
  }
}

// ---------------------------------------------------------------- bf16 WMMA GEMM
// out[b][m][n] = sum_k A[m][k] * Bt[b][n][k] + bias[m]   (M=256, N=4096, K=256)
// Double-buffered LDS tiles, async DMA pipelined against WMMA.
// 256 threads = 8 waves; block tile 64x128; wave tile 16x64; K-step 32.
__global__ __launch_bounds__(256)
void k_gemm(const unsigned short* __restrict__ A,    // [256][256] bf16
            const unsigned short* __restrict__ Bt,   // [B][4096][256] bf16
            const float* __restrict__ bias,          // [256]
            unsigned short* __restrict__ obf,        // bf16 out or null
            float* __restrict__ of32,                // fp32 out (residual path) or null
            const float* __restrict__ resid,         // residual x or null
            int transOut) {                          // bf16 out layout: [bh][L][64]
  const int tid  = threadIdx.x;
  const int lane = tid & 31;
  const int wave = tid >> 5;
  const int wm = wave & 3;                 // 4 wave-rows of 16
  const int wn = wave >> 2;                // 2 wave-cols of 64
  const int lh = lane >> 4, ll = lane & 15;
  const int mBase = blockIdx.y * 64;
  const int nBase = blockIdx.x * 128;
  const long bOff = (long)blockIdx.z * C_ * L_;     // element offset, both layouts

  __shared__ __align__(16) unsigned short lA[2][64 * 40];    // [m][k] pad 40
  __shared__ __align__(16) unsigned short lB[2][128 * 40];   // [n][k] pad 40

  f32x8 acc[4];
  #pragma unroll
  for (int t = 0; t < 4; ++t)
    #pragma unroll
    for (int j = 0; j < 8; ++j) acc[t][j] = 0.f;

  const int arow = tid >> 2, akc = tid & 3;               // A: 64 rows x 4 chunks
  const uint64_t Ab  = (uint64_t)(uintptr_t)A;
  const uint64_t Btb = (uint64_t)(uintptr_t)Bt;

  // 3 wave-level async instructions per tile stage
  auto stage = [&](int kb, int buf) {
    async_b128(lds_off32(&lA[buf][arow * 40 + akc * 8]), Ab,
               (uint32_t)(((mBase + arow) * 256 + kb + akc * 8) * 2));
    #pragma unroll
    for (int r = 0; r < 2; ++r) {
      const int fl = tid + r * 256;
      const int nrow = fl >> 2, kc = fl & 3;
      async_b128(lds_off32(&lB[buf][nrow * 40 + kc * 8]), Btb,
                 (uint32_t)((bOff + (long)(nBase + nrow) * 256 + kb + kc * 8) * 2));
    }
  };

  stage(0, 0);                               // prologue
  for (int kb = 0; kb < 256; kb += 32) {
    const int buf = (kb >> 5) & 1;
    if (kb + 32 < 256) {
      stage(kb + 32, buf ^ 1);               // next tile streams behind compute
      wait_async<3>();                       // current tile (3 oldest) complete
    } else {
      wait_async<0>();
    }
    if (kb + 64 < 256)                       // prefetch 2 tiles ahead, high locality
      __builtin_prefetch(&Bt[bOff + (long)(nBase + (tid >> 1)) * 256 + kb + 64], 0, 3);
    __syncthreads();                         // tile visible to all waves

    // A fragment: M = wm*16 + ll, K chunks at lh*8 and 16+lh*8
    Frag16 fa;
    const unsigned short* ap = &lA[buf][(wm * 16 + ll) * 40 + lh * 8];
    fa.q[0] = *(const uint4*)(ap);
    fa.q[1] = *(const uint4*)(ap + 16);
    #pragma unroll
    for (int t = 0; t < 4; ++t) {
      // B fragment: N = wn*64 + t*16 + ll, K chunks at lh*8 and 16+lh*8
      Frag16 fb;
      const unsigned short* bp = &lB[buf][(wn * 64 + t * 16 + ll) * 40 + lh * 8];
      fb.q[0] = *(const uint4*)(bp);
      fb.q[1] = *(const uint4*)(bp + 16);
      acc[t] = __builtin_amdgcn_wmma_f32_16x16x32_bf16(
          false, fa.v, false, fb.v, (short)0, acc[t], false, false);
    }
    __syncthreads();                         // reads done before buf is re-targeted
  }

  // epilogue (uniform branch): D layout M=(lh*8+j), N=ll within each 16x16 tile
  if (of32) {
    #pragma unroll
    for (int t = 0; t < 4; ++t) {
      const int nCol = nBase + wn * 64 + t * 16 + ll;
      #pragma unroll
      for (int j = 0; j < 8; ++j) {
        const int mRow = mBase + wm * 16 + lh * 8 + j;
        const long oi = bOff + (long)mRow * L_ + nCol;
        of32[oi] = acc[t][j] + bias[mRow] + resid[oi];
      }
    }
  } else if (transOut) {
    // [bh][l][c_head] layout: bh = b*NH + m/64, c = m%64
    #pragma unroll
    for (int t = 0; t < 4; ++t) {
      const int nCol = nBase + wn * 64 + t * 16 + ll;
      #pragma unroll
      for (int j = 0; j < 8; ++j) {
        const int mRow = mBase + wm * 16 + lh * 8 + j;
        const long oi = bOff + (long)(mRow >> 6) * L_ * HD_ + (long)nCol * HD_ + (mRow & 63);
        obf[oi] = f2bf(acc[t][j] + bias[mRow]);
      }
    }
  } else {
    #pragma unroll
    for (int t = 0; t < 4; ++t) {
      const int nCol = nBase + wn * 64 + t * 16 + ll;
      #pragma unroll
      for (int j = 0; j < 8; ++j) {
        const int mRow = mBase + wm * 16 + lh * 8 + j;
        obf[bOff + (long)mRow * L_ + nCol] = f2bf(acc[t][j] + bias[mRow]);
      }
    }
  }
}

// ---------------------------------------------------------------- flash attention
// grid (L/64, NH, B); 128 threads = 4 waves; each wave owns 16 query rows.
// q_t,k_t: bf16 [B*NH][L][64]   v: bf16 [B*NH][64][L]   out: bf16 ao_t[B][L][C]
__global__ __launch_bounds__(128)
void k_attn(const unsigned short* __restrict__ qt,
            const unsigned short* __restrict__ kt,
            const unsigned short* __restrict__ vg,
            unsigned short* __restrict__ aot) {
  const int tid = threadIdx.x, lane = tid & 31, wave = tid >> 5;
  const int lh = lane >> 4, ll = lane & 15;
  const int head = blockIdx.y, b = blockIdx.z;
  const long hOff = ((long)(b * NH_ + head)) * HD_ * L_;
  const int dBase = blockIdx.x * 64;
  const unsigned short* qh = qt + hOff;    // [L][64]
  const unsigned short* kh = kt + hOff;    // [L][64]
  const unsigned short* vh = vg + hOff;    // [64][L]

  __shared__ __align__(16) unsigned short kT[2][64 * 72];  // [e_local][c], double buffered
  __shared__ __align__(16) unsigned short lP[4][16 * 72];  // per-wave P tile [m][e_local]
  const uint64_t khb = (uint64_t)(uintptr_t)kh;

  // 4 wave-level async instructions per k-tile stage
  auto stageK = [&](int eb, int buf) {
    #pragma unroll
    for (int r = 0; r < 4; ++r) {
      const int fl = tid + r * 128;
      const int er = fl >> 3, cc = fl & 7;
      async_b128(lds_off32(&kT[buf][er * 72 + cc * 8]), khb,
                 (uint32_t)(((long)(eb + er) * HD_ + cc * 8) * 2));
    }
  };

  stageK(0, 0);                              // prologue (overlaps q-fragment loads)

  // q fragments (A-matrix: M=d rows of this wave, K=c) straight from global b128
  Frag16 aQ[2];
  #pragma unroll
  for (int ks = 0; ks < 2; ++ks) {
    const unsigned short* p = qh + (long)(dBase + wave * 16 + ll) * HD_ + ks * 32 + lh * 8;
    aQ[ks].q[0] = *(const uint4*)(p);
    aQ[ks].q[1] = *(const uint4*)(p + 16);
  }

  float rowmax[8], rowsum[8], alpha[8];
  f32x8 accO[4];
  #pragma unroll
  for (int j = 0; j < 8; ++j) { rowmax[j] = -3.4e38f; rowsum[j] = 0.f; }
  #pragma unroll
  for (int t = 0; t < 4; ++t)
    #pragma unroll
    for (int j = 0; j < 8; ++j) accO[t][j] = 0.f;

  for (int eb = 0; eb < L_; eb += 64) {
    const int buf = (eb >> 6) & 1;
    if (eb + 64 < L_) {
      stageK(eb + 64, buf ^ 1);              // next k-tile streams behind compute
      wait_async<4>();                       // current k-tile (4 oldest) complete
    } else {
      wait_async<0>();
    }
    __syncthreads();                         // k-tile visible to all waves

    // S = (q^T k) * scale : four 16x16 tiles over e, K=64 in two steps
    f32x8 s[4];
    #pragma unroll
    for (int t = 0; t < 4; ++t)
      #pragma unroll
      for (int j = 0; j < 8; ++j) s[t][j] = 0.f;
    #pragma unroll
    for (int ks = 0; ks < 2; ++ks) {
      #pragma unroll
      for (int t = 0; t < 4; ++t) {
        Frag16 fb;                           // B: N=e_local, K=c
        const unsigned short* p = &kT[buf][(t * 16 + ll) * 72 + ks * 32 + lh * 8];
        fb.q[0] = *(const uint4*)(p);
        fb.q[1] = *(const uint4*)(p + 16);
        s[t] = __builtin_amdgcn_wmma_f32_16x16x32_bf16(
            false, aQ[ks].v, false, fb.v, (short)0, s[t], false, false);
      }
    }
    #pragma unroll
    for (int t = 0; t < 4; ++t)
      #pragma unroll
      for (int j = 0; j < 8; ++j) s[t][j] *= 0.125f;   // hd^-0.5

    // online softmax; row m = lh*8+j, reduce over the 16 lanes sharing lh
    #pragma unroll
    for (int j = 0; j < 8; ++j) {
      float mx = fmaxf(fmaxf(s[0][j], s[1][j]), fmaxf(s[2][j], s[3][j]));
      #pragma unroll
      for (int m = 1; m <= 8; m <<= 1) mx = fmaxf(mx, __shfl_xor(mx, m, 32));
      float nm = fmaxf(rowmax[j], mx);
      float al = __expf(rowmax[j] - nm);
      rowmax[j] = nm;
      float ps = 0.f;
      #pragma unroll
      for (int t = 0; t < 4; ++t) { float e = __expf(s[t][j] - nm); s[t][j] = e; ps += e; }
      #pragma unroll
      for (int m = 1; m <= 8; m <<= 1) ps += __shfl_xor(ps, m, 32);
      rowsum[j] = rowsum[j] * al + ps;
      alpha[j] = al;
    }
    #pragma unroll
    for (int t = 0; t < 4; ++t)
      #pragma unroll
      for (int j = 0; j < 8; ++j) accO[t][j] *= alpha[j];

    // P (D-layout) -> LDS -> A-layout fragments
    #pragma unroll
    for (int t = 0; t < 4; ++t)
      #pragma unroll
      for (int j = 0; j < 8; ++j)
        lP[wave][(lh * 8 + j) * 72 + t * 16 + ll] = f2bf(s[t][j]);
    __syncthreads();   // lP visible; also fences kT[buf] reads before its next DMA

    // O += P @ V^T : A = P[m][e], B[k=e][n=c] = v[c][e] (global is already B^T-shaped)
    #pragma unroll
    for (int ks = 0; ks < 2; ++ks) {
      Frag16 fa;
      const unsigned short* p = &lP[wave][ll * 72 + ks * 32 + lh * 8];
      fa.q[0] = *(const uint4*)(p);
      fa.q[1] = *(const uint4*)(p + 16);
      #pragma unroll
      for (int t = 0; t < 4; ++t) {
        Frag16 fb;
        const unsigned short* vp = &vh[(long)(t * 16 + ll) * L_ + eb + ks * 32 + lh * 8];
        fb.q[0] = *(const uint4*)(vp);
        fb.q[1] = *(const uint4*)(vp + 16);
        accO[t] = __builtin_amdgcn_wmma_f32_16x16x32_bf16(
            false, fa.v, false, fb.v, (short)0, accO[t], false, false);
      }
    }
  }

  // normalize; write ao_t[b][l][c] ([N][K] layout for the projection GEMM)
  #pragma unroll
  for (int t = 0; t < 4; ++t) {
    const int ch = head * HD_ + t * 16 + ll;
    #pragma unroll
    for (int j = 0; j < 8; ++j) {
      const int d = dBase + wave * 16 + lh * 8 + j;
      aot[((long)b * L_ + d) * C_ + ch] = f2bf(accO[t][j] / rowsum[j]);
    }
  }
}

// ---------------------------------------------------------------- launch
extern "C" void kernel_launch(void* const* d_in, const int* in_sizes, int n_in,
                              void* d_out, int out_size, void* d_ws, size_t ws_size,
                              hipStream_t stream) {
  (void)in_sizes; (void)n_in; (void)out_size; (void)ws_size;
  const float* x    = (const float*)d_in[0];
  const float* gn_w = (const float*)d_in[1];
  const float* gn_b = (const float*)d_in[2];
  const float* wq = (const float*)d_in[3]; const float* bq = (const float*)d_in[4];
  const float* wk = (const float*)d_in[5]; const float* bk = (const float*)d_in[6];
  const float* wv = (const float*)d_in[7]; const float* bv = (const float*)d_in[8];
  const float* wp = (const float*)d_in[9]; const float* bp = (const float*)d_in[10];
  float* out = (float*)d_out;

  char* ws = (char*)d_ws;
  const size_t WB = (size_t)C_ * C_ * 2;        // 128 KiB per weight matrix
  const size_t HB = (size_t)B_ * C_ * L_ * 2;   // 8 MiB per activation tensor
  unsigned short* wq_bf = (unsigned short*)(ws);
  unsigned short* wk_bf = (unsigned short*)(ws + WB);
  unsigned short* wv_bf = (unsigned short*)(ws + 2 * WB);
  unsigned short* wp_bf = (unsigned short*)(ws + 3 * WB);
  unsigned short* h_t   = (unsigned short*)(ws + 4 * WB);            // [B][L][C]
  unsigned short* q_t   = (unsigned short*)(ws + 4 * WB + 1 * HB);   // [B*NH][L][64]
  unsigned short* k_t   = (unsigned short*)(ws + 4 * WB + 2 * HB);   // [B*NH][L][64]
  unsigned short* v_n   = (unsigned short*)(ws + 4 * WB + 3 * HB);   // [B*NH][64][L]
  unsigned short* ao_t  = (unsigned short*)(ws + 4 * WB + 4 * HB);   // [B][L][C]

  const int nW = C_ * C_;
  k_f32_to_bf16<<<(nW + 255) / 256, 256, 0, stream>>>(wq, wq_bf, nW);
  k_f32_to_bf16<<<(nW + 255) / 256, 256, 0, stream>>>(wk, wk_bf, nW);
  k_f32_to_bf16<<<(nW + 255) / 256, 256, 0, stream>>>(wv, wv_bf, nW);
  k_f32_to_bf16<<<(nW + 255) / 256, 256, 0, stream>>>(wp, wp_bf, nW);

  k_groupnorm<<<B_ * G_, 256, 0, stream>>>(x, gn_w, gn_b, h_t);

  dim3 gg(L_ / 128, C_ / 64, B_);
  k_gemm<<<gg, 256, 0, stream>>>(wq_bf, h_t, bq, q_t, nullptr, nullptr, 1);
  k_gemm<<<gg, 256, 0, stream>>>(wk_bf, h_t, bk, k_t, nullptr, nullptr, 1);
  k_gemm<<<gg, 256, 0, stream>>>(wv_bf, h_t, bv, v_n, nullptr, nullptr, 0);

  dim3 ga(L_ / 64, NH_, B_);
  k_attn<<<ga, 128, 0, stream>>>(q_t, k_t, v_n, ao_t);

  // projection + residual, fp32 out
  k_gemm<<<gg, 256, 0, stream>>>(wp_bf, ao_t, bp, nullptr, out, x, 0);
}
